// SO2NodeUpdate_27444841021658
// MI455X (gfx1250) — compile-verified, matrix operand order checked by reading, not run on hardware
//
#include <hip/hip_runtime.h>

typedef unsigned int   u32;
typedef unsigned short u16;
typedef __bf16 bf16_t;
typedef bf16_t v16bf __attribute__((ext_vector_type(16)));
typedef bf16_t v8bf  __attribute__((ext_vector_type(8)));
typedef float  v8f   __attribute__((ext_vector_type(8)));

// ---- model constants (LMAX=4, MMAX=2) ----
// perm order rows: m=0 (l=0..4), m=+1 (l=1..4), m=-1 (l=1..4), m=+2 (l=2..4), m=-2 (l=2..4)
__device__ __constant__ int c_PERM[19] = {0,2,6,11,16, 3,7,12,17, 1,5,10,15, 8,13,18, 4,9,14};
__device__ __constant__ int c_RADB[19] = {0,192,384,576,768, 960,1152,1344,1536, 960,1152,1344,1536,
                                          1728,1920,2112, 1728,1920,2112};
__device__ __constant__ int c_LROW[19] = {0,1,2,3,4, 1,2,3,4, 1,2,3,4, 2,3,4, 2,3,4};
__device__ __constant__ int c_LK[25]   = {0, 1,1,1, 2,2,2,2,2, 3,3,3,3,3, 4,4,4,4,4};

__device__ inline u16 f2bf(float f) {
  u32 u = __float_as_uint(f);
  u32 r = u + 0x7FFFu + ((u >> 16) & 1u);
  return (u16)(r >> 16);
}
__device__ inline float bf2f(u16 h) { return __uint_as_float(((u32)h) << 16); }
__device__ inline float sigmoidf_(float x) { return 1.f / (1.f + __expf(-x)); }

// monotone float<->uint map for atomicMax-based segment max
__device__ inline u32 fenc(float f) { u32 u = __float_as_uint(f); return (u & 0x80000000u) ? ~u : (u | 0x80000000u); }
__device__ inline float fdec(u32 u) { u32 v = (u & 0x80000000u) ? (u & 0x7FFFFFFFu) : ~u; return __uint_as_float(v); }

// =====================================================================
// Generic bf16 GEMM:  C[M,N] = A[M,K] (bf16, row mapped) * B[K,N] (bf16) + bias
// A row r -> edge e = r / rpe, part p = r % rpe, addr = Abase + e*sAe + p*sAp
// 128x64 output tile per block; 4 waves, each wave a 32x64 strip
// (2 A-frags x 4 B-frags = 8 v_wmma_f32_16x16x32_bf16 per wave per k-step).
// A tile staged with global_load_async_to_lds_b128 (ASYNCcnt),
// B tile staged transposed via VALU path.
// =====================================================================
__global__ __launch_bounds__(128) void gemm_bf16_kernel(
    const u16* __restrict__ Abase, long sAe, long sAp, int rpe,
    const u16* __restrict__ B, const float* __restrict__ bias,
    void* __restrict__ Cout, int storeBf16,
    int M, int N, int K)
{
  __shared__ __align__(16) u16 As[128][32];   // [row][k]
  __shared__ __align__(16) u16 Bs[64][32];    // [n][k]  (transposed tile)

  const int tid  = threadIdx.x;
  const int lane = tid & 31;
  const int wave = tid >> 5;
  const int rowBlk = blockIdx.x * 128;
  const int colBlk = blockIdx.y * 64;

  v8f zero;
#pragma unroll
  for (int i = 0; i < 8; ++i) zero[i] = 0.f;
  v8f acc[2][4];
#pragma unroll
  for (int mf = 0; mf < 2; ++mf)
#pragma unroll
    for (int nf = 0; nf < 4; ++nf) acc[mf][nf] = zero;

  // zero-fill rows beyond M once; async loads below never touch them
  if (rowBlk + tid >= M) {
    uint4 z = make_uint4(0u, 0u, 0u, 0u);
    *(uint4*)&As[tid][0]  = z;
    *(uint4*)&As[tid][8]  = z;
    *(uint4*)&As[tid][16] = z;
    *(uint4*)&As[tid][24] = z;
  }

  // per-thread A row mapping (row = rowBlk + tid)
  const int  gr  = rowBlk + tid;
  const long e_  = (long)gr / rpe;
  const long p_  = (long)gr - e_ * rpe;
  const u16* arow_ptr = Abase + e_ * sAe + p_ * sAp;
  const u32  ldsA = (u32)(uintptr_t)&As[tid][0];

  for (int k0 = 0; k0 < K; k0 += 32) {
    // ---- stage A tile (128x32) via async global->LDS copies ----
    if (gr < M) {
      const u16* rp = arow_ptr + k0;
#pragma unroll
      for (int c = 0; c < 4; ++c) {
        asm volatile("global_load_async_to_lds_b128 %0, %1, off"
                     :: "v"(ldsA + c * 16),
                        "v"((unsigned long long)(uintptr_t)(rp + c * 8))
                     : "memory");
      }
    }
    // ---- stage B tile (32x64) transposed into Bs[n][k] ----
    {
      int k   = tid & 31;
      int seg = tid >> 5;
      const u16* src = B + (long)(k0 + k) * N + colBlk + seg * 16;
      u16 v[16];
      *(uint4*)&v[0] = *(const uint4*)src;
      *(uint4*)&v[8] = *(const uint4*)(src + 8);
#pragma unroll
      for (int i = 0; i < 16; ++i) Bs[seg * 16 + i][k] = v[i];
    }
    asm volatile("s_wait_asynccnt 0" ::: "memory");
    __syncthreads();

    // ---- fragments + WMMA ----
    const int kb    = (lane < 16) ? 0 : 8;
    const int arow0 = (wave << 5) | (lane & 15);

    v8bf a0lo = *(const v8bf*)&As[arow0][kb];
    v8bf a0hi = *(const v8bf*)&As[arow0][kb + 16];
    v16bf a0 = __builtin_shufflevector(a0lo, a0hi, 0,1,2,3,4,5,6,7,8,9,10,11,12,13,14,15);
    v8bf a1lo = *(const v8bf*)&As[arow0 + 16][kb];
    v8bf a1hi = *(const v8bf*)&As[arow0 + 16][kb + 16];
    v16bf a1 = __builtin_shufflevector(a1lo, a1hi, 0,1,2,3,4,5,6,7,8,9,10,11,12,13,14,15);

#pragma unroll
    for (int nf = 0; nf < 4; ++nf) {
      int bcol = nf * 16 + (lane & 15);
      v8bf blo = *(const v8bf*)&Bs[bcol][kb];
      v8bf bhi = *(const v8bf*)&Bs[bcol][kb + 16];
      v16bf bfrag = __builtin_shufflevector(blo, bhi, 0,1,2,3,4,5,6,7,8,9,10,11,12,13,14,15);
      acc[0][nf] = __builtin_amdgcn_wmma_f32_16x16x32_bf16(
          false, a0, false, bfrag, (short)0, acc[0][nf], false, false);
      acc[1][nf] = __builtin_amdgcn_wmma_f32_16x16x32_bf16(
          false, a1, false, bfrag, (short)0, acc[1][nf], false, false);
    }
    __syncthreads();
  }

  // ---- store: lane holds col (lane&15); VGPR v -> row v + 8*(lane>=16) ----
  const int nlocal = lane & 15;
  const int rhalf  = (lane >> 4) * 8;
#pragma unroll
  for (int mf = 0; mf < 2; ++mf) {
#pragma unroll
    for (int nf = 0; nf < 4; ++nf) {
      int col = colBlk + nf * 16 + nlocal;
      float bv = bias ? bias[col] : 0.f;
#pragma unroll
      for (int v = 0; v < 8; ++v) {
        int grow = rowBlk + (wave << 5) + mf * 16 + rhalf + v;
        if (grow < M) {
          float val = acc[mf][nf][v] + bv;
          if (storeBf16) ((u16*)Cout)[(long)grow * N + col] = f2bf(val);
          else           ((float*)Cout)[(long)grow * N + col] = val;
        }
      }
    }
  }
}

// =====================================================================
// elementwise / staging kernels
// =====================================================================
__global__ void f32_to_bf16_kernel(const float* __restrict__ in, u16* __restrict__ out, long n) {
  long i = (long)blockIdx.x * blockDim.x + threadIdx.x;
  if (i < n) out[i] = f2bf(in[i]);
}

__global__ void fill_u32_kernel(u32* __restrict__ p, u32 v, long n) {
  long i = (long)blockIdx.x * blockDim.x + threadIdx.x;
  if (i < n) p[i] = v;
}

// x_edge = [edge_distance | src_embed[an[src]] | tgt_embed[an[dst]]]  -> bf16 (E,192)
__global__ void build_xedge_kernel(const float* __restrict__ edist,
                                   const float* __restrict__ semb,
                                   const float* __restrict__ temb,
                                   const int* __restrict__ anum,
                                   const int* __restrict__ eidx,
                                   u16* __restrict__ out, long E) {
  long i = (long)blockIdx.x * blockDim.x + threadIdx.x;
  if (i >= E * 192) return;
  long e = i / 192;
  int  c = (int)(i - e * 192);
  float v;
  if (c < 64)       v = edist[e * 64 + c];
  else if (c < 128) v = semb[(long)anum[eidx[e]] * 64 + (c - 64)];
  else              v = temb[(long)anum[eidx[E + e]] * 64 + (c - 128)];
  out[i] = f2bf(v);
}

// per-row LayerNorm(64) + SiLU -> bf16
__global__ void ln_silu_kernel(const float* __restrict__ in, const float* __restrict__ g,
                               const float* __restrict__ be, u16* __restrict__ out, long rows) {
  long r = (long)blockIdx.x * blockDim.x + threadIdx.x;
  if (r >= rows) return;
  const float* p = in + r * 64;
  float mu = 0.f;
#pragma unroll
  for (int i = 0; i < 64; ++i) mu += p[i];
  mu *= (1.f / 64.f);
  float var = 0.f;
#pragma unroll
  for (int i = 0; i < 64; ++i) { float d = p[i] - mu; var += d * d; }
  var *= (1.f / 64.f);
  float rs = rsqrtf(var + 1e-5f);
  u16* q = out + r * 64;
#pragma unroll
  for (int i = 0; i < 64; ++i) {
    float y = (p[i] - mu) * rs * g[i] + be[i];
    q[i] = f2bf(y * sigmoidf_(y));
  }
}

// wigner rotation fused with perm + rad scaling: msg_s[e,r,c] (bf16, perm order)
//   = rad[e, RADB[r]+c] * sum_k wigner[e, PERM[r], k] * cat[e,k,c]
// cat = [x[src] | x[dst] | edge_fea]
__global__ __launch_bounds__(256) void wigner_msg_kernel(
    const float* __restrict__ x, const float* __restrict__ efea,
    const float* __restrict__ wig, const u16* __restrict__ rad,
    const int* __restrict__ eidx, u16* __restrict__ msg_s, long E)
{
  long e = blockIdx.x;
  __shared__ float cat[25][192];
  __shared__ float wg[19][25];
  int src = eidx[e];
  for (int i = threadIdx.x; i < 25 * 192; i += 256) {
    int k = i / 192, c = i - k * 192;
    float v;
    if (c < 64)       v = x[(long)src * 1600 + k * 64 + c];
    else if (c < 128) v = x[(long)eidx[E + e] * 1600 + k * 64 + (c - 64)];
    else              v = efea[e * 1600 + k * 64 + (c - 128)];
    cat[k][c] = v;
  }
  for (int i = threadIdx.x; i < 19 * 25; i += 256) {
    int r = i / 25, k = i - r * 25;
    wg[r][k] = wig[e * 475 + (long)c_PERM[r] * 25 + k];
  }
  __syncthreads();
  for (int i = threadIdx.x; i < 19 * 192; i += 256) {
    int r = i / 192, c = i - r * 192;
    float s = 0.f;
#pragma unroll
    for (int k = 0; k < 25; ++k) s += wg[r][k] * cat[k][c];
    float scale = bf2f(rad[e * 2304 + c_RADB[r] + c]);
    msg_s[e * 3648 + i] = f2bf(s * scale);
  }
}

// conv1 complex combine + gating -> msg2 bf16 (E,19,64) in perm order
__global__ void gate_kernel(const float* __restrict__ y0, const float* __restrict__ t1,
                            const float* __restrict__ t2, u16* __restrict__ msg2, long E) {
  long i = (long)blockIdx.x * blockDim.x + threadIdx.x;
  if (i >= E * 1216) return;
  long e = i / 1216;
  int  rc = (int)(i - e * 1216);
  int  r = rc >> 6, c = rc & 63;
  const float* y0e = y0 + e * 832;
  float v;
  if (r < 5) {
    v = y0e[512 + r * 64 + c];
  } else if (r < 13) {
    int j = (r < 9) ? r - 5 : r - 9;
    const float* a = t1 + (2 * e) * 512;      // real-input row
    const float* b = t1 + (2 * e + 1) * 512;  // imag-input row
    v = (r < 9) ? (a[j * 64 + c] - b[256 + j * 64 + c])
                : (b[j * 64 + c] + a[256 + j * 64 + c]);
  } else {
    int j = (r < 16) ? r - 13 : r - 16;
    const float* a = t2 + (2 * e) * 384;
    const float* b = t2 + (2 * e + 1) * 384;
    v = (r < 16) ? (a[j * 64 + c] - b[192 + j * 64 + c])
                 : (b[j * 64 + c] + a[192 + j * 64 + c]);
  }
  if (r == 0) {
    v = v * sigmoidf_(v);                     // silu on l=0
  } else {
    int l = c_LROW[r];
    v *= sigmoidf_(y0e[256 + (l - 1) * 64 + c]);
  }
  msg2[i] = f2bf(v);
}

// per (edge, head): LayerNorm(32) + smooth_leaky_relu + dot(alpha_dot[h]) -> logits
__global__ void alpha_logits_kernel(const float* __restrict__ y0, const float* __restrict__ lng,
                                    const float* __restrict__ lnb, const float* __restrict__ adot,
                                    float* __restrict__ logits, long E) {
  long i = (long)blockIdx.x * blockDim.x + threadIdx.x;
  if (i >= E * 8) return;
  long e = i >> 3;
  int  h = (int)(i & 7);
  const float* a = y0 + e * 832 + h * 32;
  float mu = 0.f;
#pragma unroll
  for (int c = 0; c < 32; ++c) mu += a[c];
  mu *= (1.f / 32.f);
  float var = 0.f;
#pragma unroll
  for (int c = 0; c < 32; ++c) { float d = a[c] - mu; var += d * d; }
  var *= (1.f / 32.f);
  float rs = rsqrtf(var + 1e-5f);
  float s = 0.f;
#pragma unroll
  for (int c = 0; c < 32; ++c) {
    float v = (a[c] - mu) * rs * lng[c] + lnb[c];
    float act = 0.6f * v + 0.4f * v * (2.f * sigmoidf_(v) - 1.f);
    s += act * adot[h * 32 + c];
  }
  logits[i] = s;
}

__global__ void seg_max_kernel(const float* __restrict__ logits, const int* __restrict__ eidx,
                               u32* __restrict__ amax, long E) {
  long i = (long)blockIdx.x * blockDim.x + threadIdx.x;
  if (i >= E * 8) return;
  long e = i >> 3; int h = (int)(i & 7);
  atomicMax(&amax[(long)eidx[E + e] * 8 + h], fenc(logits[i]));
}

__global__ void seg_exp_kernel(const float* __restrict__ logits, const int* __restrict__ eidx,
                               const u32* __restrict__ amax, float* __restrict__ alpha,
                               float* __restrict__ denom, long E) {
  long i = (long)blockIdx.x * blockDim.x + threadIdx.x;
  if (i >= E * 8) return;
  long e = i >> 3; int h = (int)(i & 7);
  float f = __expf(logits[i] - fdec(amax[(long)eidx[E + e] * 8 + h]));
  alpha[i] = f;
  atomicAdd(&denom[(long)eidx[E + e] * 8 + h], f);
}

__global__ void alpha_norm_kernel(float* __restrict__ alpha, const int* __restrict__ eidx,
                                  const float* __restrict__ denom, long E) {
  long i = (long)blockIdx.x * blockDim.x + threadIdx.x;
  if (i >= E * 8) return;
  long e = i >> 3; int h = (int)(i & 7);
  alpha[i] = alpha[i] / (denom[(long)eidx[E + e] * 8 + h] + 1e-16f);
}

// conv2 complex combine -> msg3 bf16 (E,19,128) perm order
__global__ void combine2_kernel(const float* __restrict__ y0, const float* __restrict__ t1,
                                const float* __restrict__ t2, u16* __restrict__ msg3, long E) {
  long i = (long)blockIdx.x * blockDim.x + threadIdx.x;
  if (i >= E * 2432) return;
  long e = i / 2432;
  int  rc = (int)(i - e * 2432);
  int  r = rc >> 7, c = rc & 127;
  float v;
  if (r < 5) {
    v = y0[e * 640 + r * 128 + c];
  } else if (r < 13) {
    int j = (r < 9) ? r - 5 : r - 9;
    const float* a = t1 + (2 * e) * 1024;
    const float* b = t1 + (2 * e + 1) * 1024;
    v = (r < 9) ? (a[j * 128 + c] - b[512 + j * 128 + c])
                : (b[j * 128 + c] + a[512 + j * 128 + c]);
  } else {
    int j = (r < 16) ? r - 13 : r - 16;
    const float* a = t2 + (2 * e) * 768;
    const float* b = t2 + (2 * e + 1) * 768;
    v = (r < 16) ? (a[j * 128 + c] - b[384 + j * 128 + c])
                 : (b[j * 128 + c] + a[384 + j * 128 + c]);
  }
  msg3[i] = f2bf(v);
}

// out_e[k,c] = sum_r wigner_inv[e,k,PERM[r]] * msg3[e,r,c] * alpha[e, c/16]; atomicAdd into node[dst]
__global__ __launch_bounds__(256) void attn_out_kernel(
    const u16* __restrict__ msg3, const float* __restrict__ wigi,
    const float* __restrict__ alpha, const int* __restrict__ eidx,
    float* __restrict__ node, long E)
{
  long e = blockIdx.x;
  __shared__ float m3[19][128];
  __shared__ float wv[25][19];
  __shared__ float als[8];
  int dst = eidx[E + e];
  if (threadIdx.x < 8) als[threadIdx.x] = alpha[e * 8 + threadIdx.x];
  __syncthreads();
  for (int i = threadIdx.x; i < 19 * 128; i += 256) {
    int r = i >> 7, c = i & 127;
    m3[r][c] = bf2f(msg3[e * 2432 + i]) * als[c >> 4];
  }
  for (int i = threadIdx.x; i < 25 * 19; i += 256) {
    int k = i / 19, r = i - k * 19;
    wv[k][r] = wigi[e * 475 + (long)k * 19 + c_PERM[r]];
  }
  __syncthreads();
  for (int i = threadIdx.x; i < 25 * 128; i += 256) {
    int k = i >> 7, c = i & 127;
    float s = 0.f;
#pragma unroll
    for (int r = 0; r < 19; ++r) s += wv[k][r] * m3[r][c];
    atomicAdd(&node[(long)dst * 3200 + i], s);
  }
}

// out[b,k,o] = sum_i node[b,k,i] * proj_w[LK[k],o,i] + (k==0)*proj_b[o]
__global__ void proj_kernel(const float* __restrict__ node, const float* __restrict__ pw,
                            const float* __restrict__ pb, float* __restrict__ out, long NN) {
  long i = (long)blockIdx.x * blockDim.x + threadIdx.x;
  if (i >= NN * 25 * 64) return;
  int  o = (int)(i & 63);
  long t = i >> 6;
  int  k = (int)(t % 25);
  long b = t / 25;
  const float* w  = pw + ((long)c_LK[k] * 64 + o) * 128;
  const float* nd = node + (b * 25 + k) * 128;
  float s = (k == 0) ? pb[o] : 0.f;
#pragma unroll 4
  for (int c = 0; c < 128; ++c) s += nd[c] * w[c];
  out[i] = s;
}

// =====================================================================
// host launcher
// =====================================================================
extern "C" void kernel_launch(void* const* d_in, const int* in_sizes, int n_in,
                              void* d_out, int out_size, void* d_ws, size_t ws_size,
                              hipStream_t stream) {
  const float* x     = (const float*)d_in[0];
  const float* edist = (const float*)d_in[1];
  const float* efea  = (const float*)d_in[2];
  const float* wig   = (const float*)d_in[3];
  const float* wigi  = (const float*)d_in[4];
  const float* semb  = (const float*)d_in[5];
  const float* temb  = (const float*)d_in[6];
  const float* rw1 = (const float*)d_in[7],  *rb1 = (const float*)d_in[8];
  const float* rg1 = (const float*)d_in[9],  *rbe1= (const float*)d_in[10];
  const float* rw2 = (const float*)d_in[11], *rb2 = (const float*)d_in[12];
  const float* rg2 = (const float*)d_in[13], *rbe2= (const float*)d_in[14];
  const float* rw3 = (const float*)d_in[15], *rb3 = (const float*)d_in[16];
  const float* c1w0= (const float*)d_in[17], *c1b0= (const float*)d_in[18];
  const float* c1w1= (const float*)d_in[19], *c1w2= (const float*)d_in[20];
  const float* c2w0= (const float*)d_in[21], *c2b0= (const float*)d_in[22];
  const float* c2w1= (const float*)d_in[23], *c2w2= (const float*)d_in[24];
  const float* lng = (const float*)d_in[25], *lnb = (const float*)d_in[26];
  const float* adot= (const float*)d_in[27];
  const float* pw  = (const float*)d_in[28], *pb  = (const float*)d_in[29];
  const int* anum = (const int*)d_in[30];
  const int* eidx = (const int*)d_in[31];

  const long E  = in_sizes[31] / 2;   // 20000
  const long NN = in_sizes[30];       // 2000

  // ---- workspace layout (with aliasing) ----
  size_t o = 0;
  auto take = [&](size_t b) { size_t cur = o; o += (b + 255) & ~(size_t)255; return cur; };
  const size_t oW1R = take(192 * 64 * 2),  oW2R = take(64 * 64 * 2),   oW3R = take(64 * 2304 * 2);
  const size_t oC1W0 = take(960 * 832 * 2), oC1W1 = take(768 * 512 * 2), oC1W2 = take(576 * 384 * 2);
  const size_t oC2W0 = take(320 * 640 * 2), oC2W1 = take(256 * 1024 * 2), oC2W2 = take(192 * 768 * 2);
  const size_t oXE  = take((size_t)E * 192 * 2);
  const size_t oHP  = take((size_t)E * 64 * 4);
  const size_t oH1  = take((size_t)E * 64 * 2);
  const size_t oH2  = take((size_t)E * 64 * 2);
  const size_t oArenaA = take((size_t)E * 2304 * 2 + (size_t)E * 3648 * 2); // rad + msg_s; later tmp1_2
  const size_t oRAD  = oArenaA;
  const size_t oMSGS = oArenaA + (size_t)E * 2304 * 2;
  const size_t oY0   = take((size_t)E * 832 * 4);                           // later y0_2
  const size_t oArenaB = take((size_t)E * 2 * 512 * 4 + (size_t)E * 2 * 384 * 4); // tmp1+tmp2; later tmp2_2
  const size_t oT1 = oArenaB;
  const size_t oT2 = oArenaB + (size_t)E * 2 * 512 * 4;
  const size_t oMSG2 = take((size_t)E * 19 * 64 * 2);
  const size_t oMSG3 = take((size_t)E * 19 * 128 * 2);
  const size_t oLOG  = take((size_t)E * 8 * 4);
  const size_t oAMAX = take((size_t)NN * 8 * 4);
  const size_t oDEN  = take((size_t)NN * 8 * 4);
  const size_t oALP  = take((size_t)E * 8 * 4);
  const size_t oNODE = take((size_t)NN * 25 * 128 * 4);
  const size_t oT1_2 = oArenaA;  // E*2*1024*4 <= arenaA
  const size_t oT2_2 = oArenaB;  // E*2*768*4  <= arenaB
  const size_t oY0_2 = oY0;      // E*640*4    <= y0

  char* ws = (char*)d_ws;
  auto BFp = [&](size_t off) { return (u16*)(ws + off); };
  auto FPp = [&](size_t off) { return (float*)(ws + off); };

  auto cvt = [&](const float* src, size_t dst, long n) {
    f32_to_bf16_kernel<<<dim3((unsigned)((n + 255) / 256)), dim3(256), 0, stream>>>(src, BFp(dst), n);
  };
  auto gemm = [&](const u16* A, long sAe, long sAp, int rpe, const u16* B, const float* bias,
                  void* C, int storeBf16, long M, int N, int K) {
    dim3 grid((unsigned)((M + 127) / 128), (unsigned)(N / 64));
    gemm_bf16_kernel<<<grid, dim3(128), 0, stream>>>(A, sAe, sAp, rpe, B, bias, C, storeBf16,
                                                     (int)M, N, K);
  };

  // 0) weights -> bf16
  cvt(rw1, oW1R, 192 * 64);   cvt(rw2, oW2R, 64 * 64);     cvt(rw3, oW3R, 64 * 2304);
  cvt(c1w0, oC1W0, 960 * 832); cvt(c1w1, oC1W1, 768 * 512); cvt(c1w2, oC1W2, 576 * 384);
  cvt(c2w0, oC2W0, 320 * 640); cvt(c2w1, oC2W1, 256 * 1024); cvt(c2w2, oC2W2, 192 * 768);

  // 1) x_edge, radial MLP -> rad (bf16, E x 2304)
  build_xedge_kernel<<<dim3((unsigned)((E * 192 + 255) / 256)), dim3(256), 0, stream>>>(
      edist, semb, temb, anum, eidx, BFp(oXE), E);
  gemm(BFp(oXE), 192, 0, 1, BFp(oW1R), rb1, FPp(oHP), 0, E, 64, 192);
  ln_silu_kernel<<<dim3((unsigned)((E + 255) / 256)), dim3(256), 0, stream>>>(FPp(oHP), rg1, rbe1, BFp(oH1), E);
  gemm(BFp(oH1), 64, 0, 1, BFp(oW2R), rb2, FPp(oHP), 0, E, 64, 64);
  ln_silu_kernel<<<dim3((unsigned)((E + 255) / 256)), dim3(256), 0, stream>>>(FPp(oHP), rg2, rbe2, BFp(oH2), E);
  gemm(BFp(oH2), 64, 0, 1, BFp(oW3R), rb3, BFp(oRAD), 1, E, 2304, 64);

  // 2) wigner rotate + perm + rad scale -> msg_s (bf16, E x 19 x 192)
  wigner_msg_kernel<<<dim3((unsigned)E), dim3(256), 0, stream>>>(x, efea, wig, BFp(oRAD), eidx, BFp(oMSGS), E);

  // 3) conv1 GEMMs (WMMA)
  gemm(BFp(oMSGS),        3648, 0,   1, BFp(oC1W0), c1b0,   FPp(oY0), 0, E,     832, 960);
  gemm(BFp(oMSGS) + 960,  3648, 768, 2, BFp(oC1W1), nullptr, FPp(oT1), 0, 2 * E, 512, 768);
  gemm(BFp(oMSGS) + 2496, 3648, 576, 2, BFp(oC1W2), nullptr, FPp(oT2), 0, 2 * E, 384, 576);

  // 4) gate + combine -> msg2 ; attention logits from y0 extra
  gate_kernel<<<dim3((unsigned)((E * 1216 + 255) / 256)), dim3(256), 0, stream>>>(
      FPp(oY0), FPp(oT1), FPp(oT2), BFp(oMSG2), E);
  alpha_logits_kernel<<<dim3((unsigned)((E * 8 + 255) / 256)), dim3(256), 0, stream>>>(
      FPp(oY0), lng, lnb, adot, FPp(oLOG), E);

  // 5) segment softmax over dst
  fill_u32_kernel<<<dim3((unsigned)((NN * 8 + 255) / 256)), dim3(256), 0, stream>>>((u32*)(ws + oAMAX), 0u, NN * 8);
  fill_u32_kernel<<<dim3((unsigned)((NN * 8 + 255) / 256)), dim3(256), 0, stream>>>((u32*)(ws + oDEN), 0u, NN * 8);
  fill_u32_kernel<<<dim3((unsigned)((NN * 3200 + 255) / 256)), dim3(256), 0, stream>>>((u32*)(ws + oNODE), 0u, NN * 3200);
  seg_max_kernel<<<dim3((unsigned)((E * 8 + 255) / 256)), dim3(256), 0, stream>>>(FPp(oLOG), eidx, (u32*)(ws + oAMAX), E);
  seg_exp_kernel<<<dim3((unsigned)((E * 8 + 255) / 256)), dim3(256), 0, stream>>>(
      FPp(oLOG), eidx, (u32*)(ws + oAMAX), FPp(oALP), FPp(oDEN), E);
  alpha_norm_kernel<<<dim3((unsigned)((E * 8 + 255) / 256)), dim3(256), 0, stream>>>(FPp(oALP), eidx, FPp(oDEN), E);

  // 6) conv2 GEMMs (WMMA) — reuse arenas (msg_s/rad and tmp1/tmp2 are dead)
  gemm(BFp(oMSG2),       1216, 0,   1, BFp(oC2W0), c2b0,   FPp(oY0_2), 0, E,     640, 320);
  gemm(BFp(oMSG2) + 320, 1216, 256, 2, BFp(oC2W1), nullptr, FPp(oT1_2), 0, 2 * E, 1024, 256);
  gemm(BFp(oMSG2) + 832, 1216, 192, 2, BFp(oC2W2), nullptr, FPp(oT2_2), 0, 2 * E, 768, 192);
  combine2_kernel<<<dim3((unsigned)((E * 2432 + 255) / 256)), dim3(256), 0, stream>>>(
      FPp(oY0_2), FPp(oT1_2), FPp(oT2_2), BFp(oMSG3), E);

  // 7) alpha-weighted wigner_inv contraction + segment-sum into nodes
  attn_out_kernel<<<dim3((unsigned)E), dim3(256), 0, stream>>>(
      BFp(oMSG3), wigi, FPp(oALP), eidx, FPp(oNODE), E);

  // 8) per-l output projection
  proj_kernel<<<dim3((unsigned)((NN * 25 * 64 + 255) / 256)), dim3(256), 0, stream>>>(
      FPp(oNODE), pw, pb, (float*)d_out, NN);
}
